// TorchNeighborhoodAttention_5016521802254
// MI455X (gfx1250) — matrix operand order repaired and checked by reading, not compiled, not run on hardware
//
#include <hip/hip_runtime.h>
#include <math.h>

// ---------------------------------------------------------------------------
// Types
// ---------------------------------------------------------------------------
typedef __attribute__((ext_vector_type(16))) __bf16          bf16x16;
typedef __attribute__((ext_vector_type(16))) unsigned short  ushortx16;
typedef __attribute__((ext_vector_type(8)))  unsigned short  ushortx8;
typedef __attribute__((ext_vector_type(8)))  float           floatx8;

union FragU {
    ushortx16 u;
    bf16x16   b;
};

// Async global->LDS direct copy path (gfx1250). Guarded so both the host pass
// and toolchains lacking the builtin still compile (sync fallback).
#if defined(__AMDGCN__) && __has_builtin(__builtin_amdgcn_global_load_async_to_lds_b128)
#define USE_ASYNC_COPY 1
#else
#define USE_ASYNC_COPY 0
#endif

#if USE_ASYNC_COPY
// Payload type for the b128 variant: 4 x int vector (per compiler diagnostic).
typedef int async_b128_t __attribute__((vector_size(16)));
typedef __attribute__((address_space(1))) async_b128_t* gas_b128_ptr;
typedef __attribute__((address_space(3))) async_b128_t* las_b128_ptr;
#endif

__device__ __forceinline__ void async_copy_wait0() {
#if USE_ASYNC_COPY
#if __has_builtin(__builtin_amdgcn_s_wait_asynccnt)
    __builtin_amdgcn_s_wait_asynccnt(0);
#else
    asm volatile("s_wait_asynccnt 0" ::: "memory");
#endif
#endif
}

__device__ __forceinline__ unsigned short f32_to_bf16(float f) {
    union { float f; unsigned int u; } v;
    v.f = f;
    unsigned int r = v.u + 0x7FFFu + ((v.u >> 16) & 1u);  // round-to-nearest-even
    return (unsigned short)(r >> 16);
}

// ---------------------------------------------------------------------------
// fp32 -> bf16 conversion (vectorized, 4 elements / thread)
// ---------------------------------------------------------------------------
__global__ __launch_bounds__(256) void cvt_f32_to_bf16(
    const float* __restrict__ src, unsigned short* __restrict__ dst, int n4)
{
    int idx = blockIdx.x * blockDim.x + threadIdx.x;
    if (idx >= n4) return;
    float4 f = ((const float4*)src)[idx];
    ushort4 o;
    o.x = f32_to_bf16(f.x);
    o.y = f32_to_bf16(f.y);
    o.z = f32_to_bf16(f.z);
    o.w = f32_to_bf16(f.w);
    ((ushort4*)dst)[idx] = o;
}

// ---------------------------------------------------------------------------
// bf16 WMMA GEMM:  C[M,1024] = A[M,1024] @ W[1024,1024]^T + bias
//   A : bf16 row-major [M, K=1024]
//   W : bf16 row-major [N=1024, K=1024]  (torch Linear weight; B[k][n] = W[n][k])
//   C : fp32 row-major [M, N=1024]
// Tiling: 128x128 block tile, K-chunks of 32, 4 waves (wave32) per block.
// Each wave owns a 64x64 region = 4x4 WMMA tiles -> 16 WMMAs per K-chunk
// against 16 ds_load_b128 (4 A-frags + 4 B-frags).
// Double-buffered LDS; global->LDS via async-copy path when available.
// ---------------------------------------------------------------------------
__device__ __forceinline__ void stage_tiles(
    const unsigned short* __restrict__ A,
    const unsigned short* __restrict__ W,
    unsigned short* lA, unsigned short* lW,
    long rowBlk, int colBlk, int k0, int tid)
{
    // 128x32 bf16 tile per array = 512 x 16B vectors; 128 threads -> 4 each.
#pragma unroll
    for (int p = 0; p < 4; ++p) {
        const int vec = tid + p * 128;
        const int row = vec >> 2;
        const int seg = vec & 3;
        const unsigned short* ga = A + (rowBlk + row) * 1024 + k0 + seg * 8;
        const unsigned short* gw = W + (long)(colBlk + row) * 1024 + k0 + seg * 8;
        unsigned short* la = lA + row * 32 + seg * 8;
        unsigned short* lw = lW + row * 32 + seg * 8;
#if USE_ASYNC_COPY
        __builtin_amdgcn_global_load_async_to_lds_b128(
            (gas_b128_ptr)ga, (las_b128_ptr)la, 0, 0);
        __builtin_amdgcn_global_load_async_to_lds_b128(
            (gas_b128_ptr)gw, (las_b128_ptr)lw, 0, 0);
#else
        *(uint4*)la = *(const uint4*)ga;
        *(uint4*)lw = *(const uint4*)gw;
#endif
    }
}

__global__ __launch_bounds__(128) void gemm_bf16_wmma(
    const unsigned short* __restrict__ A,
    const unsigned short* __restrict__ W,
    const float* __restrict__ bias,
    float* __restrict__ C,
    int M)
{
    constexpr int KDIM = 1024;
    constexpr int NDIM = 1024;

    __shared__ __align__(16) unsigned short lsA[2][128 * 32];
    __shared__ __align__(16) unsigned short lsW[2][128 * 32];

    const int tid     = threadIdx.x;
    const int lane    = tid & 31;
    const int wave    = tid >> 5;
    const int warpCol = wave & 1;   // 2 column groups of 64
    const int warpRow = wave >> 1;  // 2 row groups of 64

    const long rowBlk = (long)blockIdx.y * 128;
    const int  colBlk = blockIdx.x * 128;

    floatx8 acc[4][4];
#pragma unroll
    for (int r = 0; r < 4; ++r)
#pragma unroll
        for (int c = 0; c < 4; ++c)
            acc[r][c] = (floatx8)0.0f;

    // Prime the pipeline: stage chunk 0 into buffer 0.
    stage_tiles(A, W, lsA[0], lsW[0], rowBlk, colBlk, 0, tid);
    async_copy_wait0();
    __syncthreads();

    int buf = 0;
    for (int k0 = 0; k0 < KDIM; k0 += 32) {
        const int nxt     = buf ^ 1;
        const bool hasNxt = (k0 + 32) < KDIM;

        // ---- stage next K-chunk into the other buffer (async; overlaps WMMA)
        if (hasNxt) {
            stage_tiles(A, W, lsA[nxt], lsW[nxt], rowBlk, colBlk, k0 + 32, tid);
            // prefetch the chunk after that into L2 (global_prefetch_b8)
            if (k0 + 64 < KDIM) {
                __builtin_prefetch(A + (rowBlk + (tid >> 2)) * KDIM + (k0 + 64) + (tid & 3) * 8, 0, 3);
                __builtin_prefetch(W + (long)(colBlk + (tid >> 2)) * KDIM + (k0 + 64) + (tid & 3) * 8, 0, 3);
            }
        }

        // ---- build fragments from current buffer (ISA 7.12.2 16-bit layouts)
        // A 16x32: lanes 0-15 hold row M=lane, K={0..7,16..23};
        //          lanes 16-31 hold row M=lane-16, K={8..15,24..31}
        const int koff  = (lane >> 4) * 8;
        // B 32x16: lanes 0-15 col N=lane, K={0..7,16..23}; lanes 16-31 K={8..15,24..31}
        const int kbase = (lane >> 4) * 16;

        bf16x16 afrag[4];
#pragma unroll
        for (int r = 0; r < 4; ++r) {
            const int row = warpRow * 64 + r * 16 + (lane & 15);
            ushortx8 lo = *(const ushortx8*)(&lsA[buf][row * 32 + koff]);
            ushortx8 hi = *(const ushortx8*)(&lsA[buf][row * 32 + koff + 16]);
            FragU f;
            f.u = __builtin_shufflevector(lo, hi,
                                          0, 1, 2, 3, 4, 5, 6, 7,
                                          8, 9, 10, 11, 12, 13, 14, 15);
            afrag[r] = f.b;
        }

        bf16x16 bfrag[4];
#pragma unroll
        for (int c = 0; c < 4; ++c) {
            const int col = warpCol * 64 + c * 16 + (lane & 15);
            FragU f;
            ushortx8 blo = *(const ushortx8*)(&lsW[buf][col * 32 + koff]);
            ushortx8 bhi = *(const ushortx8*)(&lsW[buf][col * 32 + koff + 16]);
            f.u = __builtin_shufflevector(blo, bhi,
                                          0, 1, 2, 3, 4, 5, 6, 7,
                                          8, 9, 10, 11, 12, 13, 14, 15);
            bfrag[c] = f.b;
        }

        // ---- 16 WMMAs per K-chunk per wave
#pragma unroll
        for (int r = 0; r < 4; ++r)
#pragma unroll
            for (int c = 0; c < 4; ++c)
                acc[r][c] = __builtin_amdgcn_wmma_f32_16x16x32_bf16(
                    false, afrag[r], false, bfrag[c],
                    (short)0, acc[r][c], false, false);

        if (hasNxt) {
            async_copy_wait0();   // next buffer's LDS writes landed (this wave)
            __syncthreads();      // ...and every wave's; also closes reads of buf
        }
        buf = nxt;
    }

    // ---- epilogue: C layout (ISA): VGPR e -> M = e (lanes 0-15) / 8+e (lanes 16-31)
#pragma unroll
    for (int r = 0; r < 4; ++r) {
#pragma unroll
        for (int c = 0; c < 4; ++c) {
            const long row0 = rowBlk + warpRow * 64 + r * 16 + ((lane >> 4) * 8);
            const int  col  = colBlk + warpCol * 64 + c * 16 + (lane & 15);
            const float bb = bias[col];
#pragma unroll
            for (int e = 0; e < 8; ++e)
                C[(row0 + e) * NDIM + col] = acc[r][c][e] + bb;
        }
    }
}

// ---------------------------------------------------------------------------
// Local (neighborhood) attention: window 7, dilation 1, HEAD_DIM=64.
// One wave per (batch, query, head); each lane holds 2 of the 64 dims.
// Scores via shfl-tree dot products, register softmax, bf16 output.
// ---------------------------------------------------------------------------
__global__ __launch_bounds__(256) void neigh_attn_kernel(
    const float* __restrict__ q,
    const float* __restrict__ k,
    const float* __restrict__ v,
    unsigned short* __restrict__ ctx)   // bf16 [B*S, 1024]
{
    constexpr int   SEQ    = 2048;
    constexpr int   HIDDEN = 1024;
    constexpr float SCALE  = 0.125f;    // 1/sqrt(64)

    const int wave = threadIdx.x >> 5;
    const int lane = threadIdx.x & 31;
    const long task = (long)blockIdx.x * (blockDim.x >> 5) + wave;  // 0 .. B*S*HEADS-1

    const int h = (int)(task & 15);
    const int i = (int)((task >> 4) & (SEQ - 1));
    const int b = (int)(task >> 15);

    const long rowBase = ((long)(b * SEQ + i) * HIDDEN) + h * 64;
    const float2 qv = *(const float2*)(q + rowBase + lane * 2);

    float s[7];
#pragma unroll
    for (int t = 0; t < 7; ++t) {
        const int j = i + t - 3;
        float sv = -__builtin_inff();
        if (j >= 0 && j < SEQ) {
            const long kBase = ((long)(b * SEQ + j) * HIDDEN) + h * 64;
            const float2 kv = *(const float2*)(k + kBase + lane * 2);
            float d = qv.x * kv.x + qv.y * kv.y;
#pragma unroll
            for (int m = 16; m >= 1; m >>= 1)
                d += __shfl_xor(d, m, 32);
            sv = d * SCALE;
        }
        s[t] = sv;
    }

    float mx = s[0];
#pragma unroll
    for (int t = 1; t < 7; ++t) mx = fmaxf(mx, s[t]);

    float p[7], sum = 0.0f;
#pragma unroll
    for (int t = 0; t < 7; ++t) {
        p[t] = __expf(s[t] - mx);   // masked entries: exp(-inf - finite) == 0
        sum += p[t];
    }
    const float inv = 1.0f / sum;

    float ox = 0.0f, oy = 0.0f;
#pragma unroll
    for (int t = 0; t < 7; ++t) {
        const int j = i + t - 3;
        if (j >= 0 && j < SEQ) {
            const long vBase = ((long)(b * SEQ + j) * HIDDEN) + h * 64;
            const float2 vv = *(const float2*)(v + vBase + lane * 2);
            ox += p[t] * vv.x;
            oy += p[t] * vv.y;
        }
    }
    ox *= inv;
    oy *= inv;

    ctx[rowBase + lane * 2 + 0] = f32_to_bf16(ox);
    ctx[rowBase + lane * 2 + 1] = f32_to_bf16(oy);
}

// ---------------------------------------------------------------------------
// Host-side launcher
// ---------------------------------------------------------------------------
extern "C" void kernel_launch(void* const* d_in, const int* in_sizes, int n_in,
                              void* d_out, int out_size, void* d_ws, size_t ws_size,
                              hipStream_t stream)
{
    constexpr int  BATCH  = 2;
    constexpr int  SEQ    = 2048;
    constexpr int  HIDDEN = 1024;
    constexpr int  MROWS  = BATCH * SEQ;                 // 4096
    constexpr long XELEMS = (long)MROWS * HIDDEN;        // 4,194,304
    constexpr long WELEMS = (long)HIDDEN * HIDDEN;       // 1,048,576

    const float* x  = (const float*)d_in[0];
    const float* Wq = (const float*)d_in[1];
    const float* bq = (const float*)d_in[2];
    const float* Wk = (const float*)d_in[3];
    const float* bk = (const float*)d_in[4];
    const float* Wv = (const float*)d_in[5];
    const float* bv = (const float*)d_in[6];
    const float* Wo = (const float*)d_in[7];
    const float* bo = (const float*)d_in[8];

    char* ws = (char*)d_ws;
    size_t off = 0;
    unsigned short* xb  = (unsigned short*)(ws + off); off += XELEMS * 2;     // 8 MB
    unsigned short* wqb = (unsigned short*)(ws + off); off += WELEMS * 2;     // 2 MB
    unsigned short* wkb = (unsigned short*)(ws + off); off += WELEMS * 2;
    unsigned short* wvb = (unsigned short*)(ws + off); off += WELEMS * 2;
    unsigned short* wob = (unsigned short*)(ws + off); off += WELEMS * 2;
    float* qf = (float*)(ws + off); off += XELEMS * 4;                        // 16 MB
    float* kf = (float*)(ws + off); off += XELEMS * 4;
    float* vf = (float*)(ws + off); off += XELEMS * 4;
    unsigned short* ctx = (unsigned short*)(ws + off); off += XELEMS * 2;     // 8 MB

    // 1) fp32 -> bf16 conversions
    {
        const int n4x = (int)(XELEMS / 4);
        const int n4w = (int)(WELEMS / 4);
        cvt_f32_to_bf16<<<(n4x + 255) / 256, 256, 0, stream>>>(x,  xb,  n4x);
        cvt_f32_to_bf16<<<(n4w + 255) / 256, 256, 0, stream>>>(Wq, wqb, n4w);
        cvt_f32_to_bf16<<<(n4w + 255) / 256, 256, 0, stream>>>(Wk, wkb, n4w);
        cvt_f32_to_bf16<<<(n4w + 255) / 256, 256, 0, stream>>>(Wv, wvb, n4w);
        cvt_f32_to_bf16<<<(n4w + 255) / 256, 256, 0, stream>>>(Wo, wob, n4w);
    }

    // 2) Q/K/V projections (bf16 WMMA GEMMs, fp32 out)
    {
        dim3 grid(HIDDEN / 128, MROWS / 128);  // (8, 32)
        gemm_bf16_wmma<<<grid, 128, 0, stream>>>(xb, wqb, bq, qf, MROWS);
        gemm_bf16_wmma<<<grid, 128, 0, stream>>>(xb, wkb, bk, kf, MROWS);
        gemm_bf16_wmma<<<grid, 128, 0, stream>>>(xb, wvb, bv, vf, MROWS);
    }

    // 3) neighborhood attention (one wave per (b, i, h)), bf16 ctx out
    {
        const long tasks  = (long)MROWS * 16;       // 65536 waves
        const int  blocks = (int)(tasks / 8);       // 8 waves per 256-thread block
        neigh_attn_kernel<<<blocks, 256, 0, stream>>>(qf, kf, vf, ctx);
    }

    // 4) output projection (bf16 WMMA GEMM, fp32 result -> d_out)
    {
        dim3 grid(HIDDEN / 128, MROWS / 128);
        gemm_bf16_wmma<<<grid, 128, 0, stream>>>(ctx, wob, bo, (float*)d_out, MROWS);
    }
}
